// SpikingPendulumBrain_40312563040421
// MI455X (gfx1250) — matrix operand order repaired and checked by reading, not compiled, not run on hardware
//
#include <hip/hip_runtime.h>
#include <hip/hip_bf16.h>

typedef __attribute__((ext_vector_type(16))) _Float16 v16h;
typedef __attribute__((ext_vector_type(8)))  _Float16 v8h;
typedef __attribute__((ext_vector_type(8)))  float    v8f;

#define NB 64          // batch
#define NN 4096        // neurons
#define STEPS 50
#define KC  256        // K-chunk staged in LDS per barrier
#define SSTRIDE 264    // LDS row stride in halves (256 + 8 pad -> 4-bank shift/row)
#define DT_TAU   0.005f            // DT / TAU_M
#define ONE_MDT  0.995f            // 1 - DT/TAU_M
#define SYN_DEC  0.98019867f       // exp(-DT/TAU_SYN)
#define V_TH     20.0f

// ---- W_rec f32 -> f16 (one pass; 33.5MB f16 stays L2-resident afterwards) ----
__global__ void cvt_wrec_kernel(const float* __restrict__ w, _Float16* __restrict__ wh) {
    size_t i = (size_t)blockIdx.x * blockDim.x + threadIdx.x;
    wh[i] = (_Float16)w[i];
}

// ---- init: i_in = 5*(obs @ W_in.T), zero state + accumulators ----
__global__ void init_kernel(const float* __restrict__ obs, const float* __restrict__ Win,
                            float* __restrict__ i_in, float* __restrict__ v,
                            float* __restrict__ isyn, _Float16* __restrict__ sp0,
                            _Float16* __restrict__ sp1, float* __restrict__ step_dot,
                            float* __restrict__ acc) {
    int idx = blockIdx.x * blockDim.x + threadIdx.x;    // 0 .. NB*NN-1
    int b = idx >> 12;
    int j = idx & (NN - 1);
    i_in[idx] = 5.0f * (obs[b * 2] * Win[j * 2] + obs[b * 2 + 1] * Win[j * 2 + 1]);
    v[idx] = 0.0f;
    isyn[idx] = 0.0f;
    sp0[idx] = (_Float16)0.0f;
    sp1[idx] = (_Float16)0.0f;
    if (idx < NB) { step_dot[idx] = 0.0f; acc[idx] = 0.0f; }
}

// ---- one brain step: i_rec GEMM via WMMA + fused LIF update + motor partials ----
__global__ void __launch_bounds__(256)
step_kernel(const _Float16* __restrict__ sp_in, _Float16* __restrict__ sp_out,
            const _Float16* __restrict__ Wh, const float* __restrict__ i_in,
            float* __restrict__ v, float* __restrict__ isyn,
            const float* __restrict__ Wout, float* __restrict__ step_dot) {
    // Spike A-panel staged in LDS: 64 rows x 256 K-halves per barrier (~34 KB)
    __shared__ __align__(16) _Float16 smem[64 * SSTRIDE];

    const int tid    = threadIdx.x;
    const int lane   = tid & 31;
    const int w      = tid >> 5;         // wave id 0..7
    const int m_tile = w & 3;            // 4 M-tiles cover batch 64
    const int n_grp  = w >> 2;           // 2 groups of 32 columns
    const int lane16 = lane & 15;
    const int hi     = lane >> 4;        // half-wave select

    const int n_base = blockIdx.x * 64 + n_grp * 32;
    const int j0 = n_base + lane16;      // first 16x16 N-subtile column
    const int j1 = j0 + 16;              // second N-subtile column

    // A fragment: row = M, 16 contiguous K at hi*16
    const int arow = m_tile * 16 + lane16;
    const int alo  = hi * 16;
    // B fragment: lane holds one W row (column of B), 16 contiguous K at hi*16
    const int bko  = hi * 16;

    // LDS staging: 256 threads x 64 halves (8 x b128) = 64x256 tile
    const int srow = tid >> 2;
    const int sseg = (tid & 3) * 64;

    const _Float16* bp0 = Wh + (size_t)j0 * NN + bko;
    const _Float16* bp1 = Wh + (size_t)j1 * NN + bko;

    v8f c0 = {}, c1 = {};

    for (int k0 = 0; k0 < NN; k0 += KC) {
        __syncthreads();
#pragma unroll
        for (int u = 0; u < 8; ++u) {
            *reinterpret_cast<v8h*>(&smem[srow * SSTRIDE + sseg + u * 8]) =
                *reinterpret_cast<const v8h*>(&sp_in[(size_t)srow * NN + k0 + sseg + u * 8]);
        }
        __syncthreads();

#pragma unroll
        for (int ks = 0; ks < KC / 32; ++ks) {
            const int kk = ks * 32;
            union { v16h v; v8h h[2]; } af, bf0, bf1;
            af.h[0]  = *reinterpret_cast<const v8h*>(&smem[arow * SSTRIDE + kk + alo]);
            af.h[1]  = *reinterpret_cast<const v8h*>(&smem[arow * SSTRIDE + kk + alo + 8]);
            bf0.h[0] = *reinterpret_cast<const v8h*>(bp0 + k0 + kk);
            bf0.h[1] = *reinterpret_cast<const v8h*>(bp0 + k0 + kk + 8);
            bf1.h[0] = *reinterpret_cast<const v8h*>(bp1 + k0 + kk);
            bf1.h[1] = *reinterpret_cast<const v8h*>(bp1 + k0 + kk + 8);

            c0 = __builtin_amdgcn_wmma_f32_16x16x32_f16(false, af.v, false, bf0.v,
                                                        (short)0, c0, false, false);
            c1 = __builtin_amdgcn_wmma_f32_16x16x32_f16(false, af.v, false, bf1.v,
                                                        (short)0, c1, false, false);
        }
    }

    // Fused LIF update + spike write + motor partial dot.
    // C/D layout: VGPR r -> M = r + 8*hi ; lane%16 -> N
    const float wo0 = Wout[j0];
    const float wo1 = Wout[j1];
    float part[8];
#pragma unroll
    for (int r = 0; r < 8; ++r) {
        const int b = m_tile * 16 + r + hi * 8;
        {
            size_t idx = (size_t)b * NN + j0;
            float is = isyn[idx] * SYN_DEC + i_in[idx] + c0[r];
            float vv = v[idx] * ONE_MDT + is * DT_TAU;
            float s  = (vv >= V_TH) ? 1.0f : 0.0f;
            v[idx]    = vv * (1.0f - s);
            isyn[idx] = is;
            sp_out[idx] = (_Float16)s;
            part[r] = s * wo0;
        }
        {
            size_t idx = (size_t)b * NN + j1;
            float is = isyn[idx] * SYN_DEC + i_in[idx] + c1[r];
            float vv = v[idx] * ONE_MDT + is * DT_TAU;
            float s  = (vv >= V_TH) ? 1.0f : 0.0f;
            v[idx]    = vv * (1.0f - s);
            isyn[idx] = is;
            sp_out[idx] = (_Float16)s;
            part[r] += s * wo1;
        }
        // reduce over the 16 lanes that share batch row b (xor stays in half-wave)
#pragma unroll
        for (int m = 1; m <= 8; m <<= 1)
            part[r] += __shfl_xor(part[r], m, 32);
    }
    if (lane16 == 0) {
#pragma unroll
        for (int r = 0; r < 8; ++r)
            atomicAdd(&step_dot[m_tile * 16 + r + hi * 8], part[r]);
    }
}

// ---- per-step motor: acc += tanh(step_dot); reset step_dot ----
__global__ void tanh_step_kernel(float* __restrict__ step_dot, float* __restrict__ acc) {
    int b = threadIdx.x;
    if (b < NB) {
        acc[b] += tanhf(step_dot[b]);
        step_dot[b] = 0.0f;
    }
}

__global__ void final_kernel(const float* __restrict__ acc, float* __restrict__ out) {
    int b = threadIdx.x;
    if (b < NB) out[b] = acc[b] * (1.0f / (float)STEPS);
}

extern "C" void kernel_launch(void* const* d_in, const int* in_sizes, int n_in,
                              void* d_out, int out_size, void* d_ws, size_t ws_size,
                              hipStream_t stream) {
    const float* obs  = (const float*)d_in[0];   // [64,2]
    const float* Win  = (const float*)d_in[1];   // [4096,2]
    const float* Wrec = (const float*)d_in[2];   // [4096,4096]
    const float* Wout = (const float*)d_in[3];   // [1,4096]
    // d_in[4] = n_brain_steps (50, fixed per reference)

    char* p = (char*)d_ws;
    _Float16* Wh    = (_Float16*)p;  p += (size_t)NN * NN * sizeof(_Float16);  // 33.5 MB
    float* i_in     = (float*)p;     p += (size_t)NB * NN * sizeof(float);
    float* v        = (float*)p;     p += (size_t)NB * NN * sizeof(float);
    float* isyn     = (float*)p;     p += (size_t)NB * NN * sizeof(float);
    _Float16* sp0   = (_Float16*)p;  p += (size_t)NB * NN * sizeof(_Float16);
    _Float16* sp1   = (_Float16*)p;  p += (size_t)NB * NN * sizeof(_Float16);
    float* step_dot = (float*)p;     p += 256;
    float* acc      = (float*)p;     p += 256;

    hipLaunchKernelGGL(cvt_wrec_kernel, dim3(((size_t)NN * NN) / 256), dim3(256), 0, stream,
                       Wrec, Wh);
    hipLaunchKernelGGL(init_kernel, dim3((NB * NN) / 256), dim3(256), 0, stream,
                       obs, Win, i_in, v, isyn, sp0, sp1, step_dot, acc);

    for (int t = 0; t < STEPS; ++t) {
        _Float16* si = (t & 1) ? sp1 : sp0;
        _Float16* so = (t & 1) ? sp0 : sp1;
        hipLaunchKernelGGL(step_kernel, dim3(NN / 64), dim3(256), 0, stream,
                           si, so, Wh, i_in, v, isyn, Wout, step_dot);
        hipLaunchKernelGGL(tanh_step_kernel, dim3(1), dim3(64), 0, stream, step_dot, acc);
    }

    hipLaunchKernelGGL(final_kernel, dim3(1), dim3(64), 0, stream, acc, (float*)d_out);
}
